// MERNNEncoder_31662498906281
// MI455X (gfx1250) — matrix operand order repaired and checked by reading, not compiled, not run on hardware
//
#include <hip/hip_runtime.h>
#include <hip/hip_bf16.h>

#define B_   32
#define T_   384
#define D_   768
#define D3_  (3 * D_)
#define BT_  (B_ * T_)
#define THRESH_ 0.95f

typedef __bf16 bf16_t;
typedef __attribute__((ext_vector_type(16))) __bf16 bf16x16;
typedef __attribute__((ext_vector_type(8)))  __bf16 bf16x8;
typedef __attribute__((ext_vector_type(8)))  float  f32x8;

// ---------------------------------------------------------------------------
// WMMA helpers (CDNA5 16x16x32 bf16 -> f32)
// A fragment: 16x32 (MxK); lanes 0-15 hold K 0..15, lanes 16-31 hold
// K 16..31 (same order as B fragment).
// ---------------------------------------------------------------------------
__device__ __forceinline__ bf16x16 load_frag_a(const bf16_t* __restrict__ base,
                                               int ld, int lane) {
    const int lg = lane >> 4;
    const int m  = lane & 15;
    const bf16_t* p = base + (long)m * ld + lg * 16;
    union { bf16x16 v; bf16x8 h[2]; } u;
    u.h[0] = *(const bf16x8*)(p);       // K = lg*16  .. lg*16+7
    u.h[1] = *(const bf16x8*)(p + 8);   // K = lg*16+8.. lg*16+15
    return u.v;
}

// B fragment: 32x16 (KxN) from row-major [N x K] weight (computes A*W^T).
// lanes 0-15: column n, K 0..15 ; lanes 16-31: column n, K 16..31.
__device__ __forceinline__ bf16x16 load_frag_b(const bf16_t* __restrict__ base,
                                               int ld, int lane) {
    const int lg = lane >> 4;
    const int n  = lane & 15;
    const bf16_t* p = base + (long)n * ld + lg * 16;
    union { bf16x16 v; bf16x8 h[2]; } u;
    u.h[0] = *(const bf16x8*)(p);
    u.h[1] = *(const bf16x8*)(p + 8);
    return u.v;
}

__device__ __forceinline__ f32x8 wmma_bf16(bf16x16 a, bf16x16 b, f32x8 c) {
    return __builtin_amdgcn_wmma_f32_16x16x32_bf16(false, a, false, b,
                                                   (short)0, c, false, false);
}

__device__ __forceinline__ float sigmoidf_(float x) {
    return 1.0f / (1.0f + __expf(-x));
}

// ---------------------------------------------------------------------------
// Prep kernels
// ---------------------------------------------------------------------------
__global__ void convert_weights_kernel(const float* __restrict__ wih,
                                       const float* __restrict__ whh,
                                       bf16_t* __restrict__ wih_b,
                                       bf16_t* __restrict__ whh_b) {
    int i = blockIdx.x * blockDim.x + threadIdx.x;
    if (i < D3_ * D_) {
        wih_b[i] = (bf16_t)wih[i];
        whh_b[i] = (bf16_t)whh[i];
    }
}

__global__ void gather_embed_kernel(const int* __restrict__ sent,
                                    const float* __restrict__ emb,
                                    bf16_t* __restrict__ x) {
    long i = (long)blockIdx.x * blockDim.x + threadIdx.x;   // over BT_*D_
    if (i < (long)BT_ * D_) {
        int row = (int)(i / D_);
        int d   = (int)(i - (long)row * D_);
        int tok = sent[row];
        x[i] = (bf16_t)emb[(long)tok * D_ + d];
    }
}

__global__ void init_state_kernel(float* __restrict__ hf, bf16_t* __restrict__ hb) {
    int i = blockIdx.x * blockDim.x + threadIdx.x;
    if (i < B_ * D_) { hf[i] = 0.0f; hb[i] = (bf16_t)0.0f; }
}

__global__ void zero_out_kernel(float* __restrict__ o, long n) {
    long i = (long)blockIdx.x * blockDim.x + threadIdx.x;
    long stride = (long)gridDim.x * blockDim.x;
    for (; i < n; i += stride) o[i] = 0.0f;
}

// ---------------------------------------------------------------------------
// xp = X(bf16)[BT_,D_] * W_ih^T(bf16)[D3_,D_] + b_ih -> fp32 [BT_, D3_]
// grid: (BT_/32, D3_/64), one wave per block; 32x64 tile per wave.
// Two-slot ping-pong pipeline (unroll-by-2): no register rotation copies,
// loads target the slot last read two k-steps ago (outside hazard window).
// ---------------------------------------------------------------------------
__global__ void __launch_bounds__(32)
__attribute__((amdgpu_waves_per_eu(1)))
gemm_xp_kernel(const bf16_t* __restrict__ X,
               const bf16_t* __restrict__ W,
               const float* __restrict__ bias,
               float* __restrict__ out) {
    const int lane = threadIdx.x;
    const int m0 = blockIdx.x * 32;
    const int n0 = blockIdx.y * 64;
    const bf16_t* Xa = X + (long)m0 * D_;
    const bf16_t* Xb = X + (long)(m0 + 16) * D_;
    const bf16_t* Wn[4];
#pragma unroll
    for (int j = 0; j < 4; ++j) Wn[j] = W + (long)(n0 + j * 16) * D_;

    f32x8 acc0[4] = {};
    f32x8 acc1[4] = {};
    bf16x16 a0[2], a1[2], bb[2][4];

    // slot 0 <- k = 0
    a0[0] = load_frag_a(Xa, D_, lane);
    a1[0] = load_frag_a(Xb, D_, lane);
#pragma unroll
    for (int j = 0; j < 4; ++j) bb[0][j] = load_frag_b(Wn[j], D_, lane);

    for (int k0 = 0; k0 < D_ - 64; k0 += 64) {
        // slot 1 <- k0+32
        a0[1] = load_frag_a(Xa + k0 + 32, D_, lane);
        a1[1] = load_frag_a(Xb + k0 + 32, D_, lane);
#pragma unroll
        for (int j = 0; j < 4; ++j) bb[1][j] = load_frag_b(Wn[j] + k0 + 32, D_, lane);
        // consume slot 0 (k0)
#pragma unroll
        for (int j = 0; j < 4; ++j) {
            acc0[j] = wmma_bf16(a0[0], bb[0][j], acc0[j]);
            acc1[j] = wmma_bf16(a1[0], bb[0][j], acc1[j]);
        }
        // slot 0 <- k0+64
        a0[0] = load_frag_a(Xa + k0 + 64, D_, lane);
        a1[0] = load_frag_a(Xb + k0 + 64, D_, lane);
#pragma unroll
        for (int j = 0; j < 4; ++j) bb[0][j] = load_frag_b(Wn[j] + k0 + 64, D_, lane);
        // consume slot 1 (k0+32)
#pragma unroll
        for (int j = 0; j < 4; ++j) {
            acc0[j] = wmma_bf16(a0[1], bb[1][j], acc0[j]);
            acc1[j] = wmma_bf16(a1[1], bb[1][j], acc1[j]);
        }
    }
    // epilogue pair: slot0 holds D_-64; slot1 <- D_-32
    a0[1] = load_frag_a(Xa + D_ - 32, D_, lane);
    a1[1] = load_frag_a(Xb + D_ - 32, D_, lane);
#pragma unroll
    for (int j = 0; j < 4; ++j) bb[1][j] = load_frag_b(Wn[j] + D_ - 32, D_, lane);
#pragma unroll
    for (int j = 0; j < 4; ++j) {
        acc0[j] = wmma_bf16(a0[0], bb[0][j], acc0[j]);
        acc1[j] = wmma_bf16(a1[0], bb[0][j], acc1[j]);
    }
#pragma unroll
    for (int j = 0; j < 4; ++j) {
        acc0[j] = wmma_bf16(a0[1], bb[1][j], acc0[j]);
        acc1[j] = wmma_bf16(a1[1], bb[1][j], acc1[j]);
    }

    const int lg = lane >> 4, nl = lane & 15;
#pragma unroll
    for (int j = 0; j < 4; ++j)
#pragma unroll
        for (int r = 0; r < 8; ++r) {
            int n = n0 + j * 16 + nl;
            int ma = m0 + r + lg * 8;
            int mb = m0 + 16 + r + lg * 8;
            out[(long)ma * D3_ + n] = acc0[j][r] + bias[n];
            out[(long)mb * D3_ + n] = acc1[j][r] + bias[n];
        }
}

// ---------------------------------------------------------------------------
// One GRU time step. grid: D_/16 = 48 blocks, one wave each.
// Block owns output columns d0..d0+15 for all 32 batch rows, all 3 gates.
// hb double-buffered across launches (cross-block RAW); hf column-owned.
// Same two-slot ping-pong K pipeline.
// ---------------------------------------------------------------------------
__global__ void __launch_bounds__(32)
__attribute__((amdgpu_waves_per_eu(1)))
gru_step_kernel(const bf16_t* __restrict__ Whh,
                const float* __restrict__ bhh,
                const float* __restrict__ xp,
                float* __restrict__ hf,
                const bf16_t* __restrict__ hb_in,
                bf16_t* __restrict__ hb_out,
                float* __restrict__ states, int t) {
    const int lane = threadIdx.x;
    const int d0 = blockIdx.x * 16;
    const bf16_t* Ha = hb_in;
    const bf16_t* Hb = hb_in + 16 * D_;
    const bf16_t* Wg[3];
#pragma unroll
    for (int g = 0; g < 3; ++g) Wg[g] = Whh + (long)(g * D_ + d0) * D_;

    f32x8 acc[3][2] = {};
    bf16x16 a0[2], a1[2], bb[2][3];

    a0[0] = load_frag_a(Ha, D_, lane);
    a1[0] = load_frag_a(Hb, D_, lane);
#pragma unroll
    for (int g = 0; g < 3; ++g) bb[0][g] = load_frag_b(Wg[g], D_, lane);

    for (int k0 = 0; k0 < D_ - 64; k0 += 64) {
        a0[1] = load_frag_a(Ha + k0 + 32, D_, lane);
        a1[1] = load_frag_a(Hb + k0 + 32, D_, lane);
#pragma unroll
        for (int g = 0; g < 3; ++g) bb[1][g] = load_frag_b(Wg[g] + k0 + 32, D_, lane);
#pragma unroll
        for (int g = 0; g < 3; ++g) {
            acc[g][0] = wmma_bf16(a0[0], bb[0][g], acc[g][0]);
            acc[g][1] = wmma_bf16(a1[0], bb[0][g], acc[g][1]);
        }
        a0[0] = load_frag_a(Ha + k0 + 64, D_, lane);
        a1[0] = load_frag_a(Hb + k0 + 64, D_, lane);
#pragma unroll
        for (int g = 0; g < 3; ++g) bb[0][g] = load_frag_b(Wg[g] + k0 + 64, D_, lane);
#pragma unroll
        for (int g = 0; g < 3; ++g) {
            acc[g][0] = wmma_bf16(a0[1], bb[1][g], acc[g][0]);
            acc[g][1] = wmma_bf16(a1[1], bb[1][g], acc[g][1]);
        }
    }
    a0[1] = load_frag_a(Ha + D_ - 32, D_, lane);
    a1[1] = load_frag_a(Hb + D_ - 32, D_, lane);
#pragma unroll
    for (int g = 0; g < 3; ++g) bb[1][g] = load_frag_b(Wg[g] + D_ - 32, D_, lane);
#pragma unroll
    for (int g = 0; g < 3; ++g) {
        acc[g][0] = wmma_bf16(a0[0], bb[0][g], acc[g][0]);
        acc[g][1] = wmma_bf16(a1[0], bb[0][g], acc[g][1]);
    }
#pragma unroll
    for (int g = 0; g < 3; ++g) {
        acc[g][0] = wmma_bf16(a0[1], bb[1][g], acc[g][0]);
        acc[g][1] = wmma_bf16(a1[1], bb[1][g], acc[g][1]);
    }

    const int lg = lane >> 4, nl = lane & 15;
    const int d = d0 + nl;
    const float bhr = bhh[d], bhz = bhh[D_ + d], bhn = bhh[2 * D_ + d];
#pragma unroll
    for (int mt = 0; mt < 2; ++mt)
#pragma unroll
        for (int r = 0; r < 8; ++r) {
            int b = mt * 16 + r + lg * 8;
            long xrow = ((long)b * T_ + t) * D3_;
            float rg = sigmoidf_(xp[xrow + d] + acc[0][mt][r] + bhr);
            float zg = sigmoidf_(xp[xrow + D_ + d] + acc[1][mt][r] + bhz);
            float ng = tanhf(xp[xrow + 2 * D_ + d] + rg * (acc[2][mt][r] + bhn));
            float hold = hf[b * D_ + d];
            float hnew = (1.0f - zg) * ng + zg * hold;
            hf[b * D_ + d] = hnew;
            hb_out[b * D_ + d] = (bf16_t)hnew;
            states[((long)b * T_ + t) * D_ + d] = hnew;
        }
}

// ---------------------------------------------------------------------------
// probs = sigmoid(states @ w_act + b_act) ; one wave per (b,t) row.
// ---------------------------------------------------------------------------
__global__ void probs_kernel(const float* __restrict__ states,
                             const float* __restrict__ wact,
                             const float* __restrict__ bact,
                             float* __restrict__ probs) {
    const int row = blockIdx.x;
    const int lane = threadIdx.x;
    const float* s = states + (long)row * D_;
    float sum = 0.0f;
    for (int d = lane; d < D_; d += 32) sum += s[d] * wact[d];
#pragma unroll
    for (int off = 16; off > 0; off >>= 1) sum += __shfl_down(sum, off, 32);
    if (lane == 0) probs[row] = sigmoidf_(sum + bact[0]);
}

// ---------------------------------------------------------------------------
// ACT halting + segment pooling. One block per batch row; threads own columns.
// weights[t]=p except halting step: p - (seg_sum - 1) => out = acc - (sum-1)*s_t
// ---------------------------------------------------------------------------
__global__ void segment_kernel(const float* __restrict__ states,
                               const float* __restrict__ probs,
                               float* __restrict__ embs) {
    const int b = blockIdx.x;
    const int tid = threadIdx.x;  // 256 threads, 3 columns each
    float acc[3] = {0.f, 0.f, 0.f};
    float sump = 0.f;
    int seg = 0;
    for (int t = 0; t < T_; ++t) {
        float p = probs[b * T_ + t];
        const float* s = states + ((long)b * T_ + t) * D_;
        float sv[3];
#pragma unroll
        for (int j = 0; j < 3; ++j) { sv[j] = s[tid + j * 256]; acc[j] += p * sv[j]; }
        float ns = sump + p;
        if (ns > THRESH_) {
            float corr = ns - 1.0f;
            float* o = embs + ((long)b * T_ + seg) * D_;
#pragma unroll
            for (int j = 0; j < 3; ++j) { o[tid + j * 256] = acc[j] - corr * sv[j]; acc[j] = 0.f; }
            sump = 0.f;
            ++seg;
        } else {
            sump = ns;
        }
    }
}

// ---------------------------------------------------------------------------
extern "C" void kernel_launch(void* const* d_in, const int* in_sizes, int n_in,
                              void* d_out, int out_size, void* d_ws, size_t ws_size,
                              hipStream_t stream) {
    (void)in_sizes; (void)n_in; (void)out_size; (void)ws_size;

    const int*   sentence = (const int*)  d_in[0];
    const float* emb      = (const float*)d_in[1];
    const float* w_ih     = (const float*)d_in[2];
    const float* w_hh     = (const float*)d_in[3];
    const float* b_ih     = (const float*)d_in[4];
    const float* b_hh     = (const float*)d_in[5];
    const float* w_act    = (const float*)d_in[6];
    const float* b_act    = (const float*)d_in[7];

    float* out_embs  = (float*)d_out;                       // [BT_, D_]
    float* out_probs = (float*)d_out + (long)BT_ * D_;      // [BT_]

    // ---- workspace carve-up (256B aligned) ----
    char* ws = (char*)d_ws;
    size_t off = 0;
    auto carve = [&](size_t bytes) -> char* {
        char* p = ws + off;
        off = (off + bytes + 255) & ~(size_t)255;
        return p;
    };
    bf16_t* x_b    = (bf16_t*)carve((size_t)BT_ * D_ * sizeof(bf16_t));
    bf16_t* wih_b  = (bf16_t*)carve((size_t)D3_ * D_ * sizeof(bf16_t));
    bf16_t* whh_b  = (bf16_t*)carve((size_t)D3_ * D_ * sizeof(bf16_t));
    float*  xp     = (float*) carve((size_t)BT_ * D3_ * sizeof(float));
    float*  h_f32  = (float*) carve((size_t)B_ * D_ * sizeof(float));
    bf16_t* h_bf   = (bf16_t*)carve((size_t)2 * B_ * D_ * sizeof(bf16_t));  // ping-pong
    float*  states = (float*) carve((size_t)BT_ * D_ * sizeof(float));

    // ---- 1. convert weights to bf16 ----
    {
        int n = D3_ * D_;
        convert_weights_kernel<<<(n + 255) / 256, 256, 0, stream>>>(w_ih, w_hh, wih_b, whh_b);
    }
    // ---- 2. embedding gather -> bf16 ----
    {
        long n = (long)BT_ * D_;
        gather_embed_kernel<<<(int)((n + 255) / 256), 256, 0, stream>>>(sentence, emb, x_b);
    }
    // ---- 3. init h = 0 (fp32 + bf16 buffer 0) ----
    init_state_kernel<<<(B_ * D_ + 255) / 256, 256, 0, stream>>>(h_f32, h_bf);
    // ---- 4. zero output embs region (incomplete segments must be 0) ----
    zero_out_kernel<<<2048, 256, 0, stream>>>(out_embs, (long)BT_ * D_);
    // ---- 5. xp = X * W_ih^T + b_ih  (WMMA bf16, 32x64 tile, ping-pong) ----
    {
        dim3 grid(BT_ / 32, D3_ / 64);
        gemm_xp_kernel<<<grid, 32, 0, stream>>>(x_b, wih_b, b_ih, xp);
    }
    // ---- 6. sequential GRU recurrence (WMMA bf16 per step, ping-pong) ----
    for (int t = 0; t < T_; ++t) {
        const bf16_t* hb_in  = h_bf + (size_t)(t & 1) * B_ * D_;
        bf16_t*       hb_out = h_bf + (size_t)((t + 1) & 1) * B_ * D_;
        gru_step_kernel<<<D_ / 16, 32, 0, stream>>>(whh_b, b_hh, xp, h_f32,
                                                    hb_in, hb_out, states, t);
    }
    // ---- 7. halting probabilities (into d_out tail) ----
    probs_kernel<<<BT_, 32, 0, stream>>>(states, w_act, b_act, out_probs);
    // ---- 8. ACT segment pooling ----
    segment_kernel<<<B_, 256, 0, stream>>>(states, out_probs, out_embs);
}
